// CollaborativeWaterfallMoE_74105365725762
// MI455X (gfx1250) — compile-verified, matrix operand order checked by reading, not run on hardware
//
#include <hip/hip_runtime.h>
#include <hip/hip_bf16.h>
#include <math.h>

typedef _Float16 h16;
typedef __attribute__((ext_vector_type(16))) _Float16 v16h;
typedef __attribute__((ext_vector_type(8)))  _Float16 v8h;
typedef __attribute__((ext_vector_type(4)))  _Float16 v4h;
typedef __attribute__((ext_vector_type(8)))  float    v8f;
typedef __attribute__((ext_vector_type(4)))  float    v4f;

#define E_EXP 4
#define BTOK  1024
#define CAPC  256
#define NCLS  10

// ---------------------------------------------------------------------------
// Weight conversion f32 -> f16  (layout [Cout][Cin*9], contiguous in K)
// ---------------------------------------------------------------------------
__global__ __launch_bounds__(256)
void convert_w(const float* __restrict__ w1, const float* __restrict__ w2,
               const float* __restrict__ w3, const float* __restrict__ w4,
               h16* __restrict__ o1, h16* __restrict__ o2,
               h16* __restrict__ o3, h16* __restrict__ o4)
{
    const int N1 = E_EXP*64*27, N2 = E_EXP*64*576, N3 = E_EXP*128*576, N4 = E_EXP*256*1152;
    int t = blockIdx.x * blockDim.x + threadIdx.x;
    if (t < N1)                { o1[t] = (h16)w1[t]; return; }
    t -= N1;
    if (t < N2)                { o2[t] = (h16)w2[t]; return; }
    t -= N2;
    if (t < N3)                { o3[t] = (h16)w3[t]; return; }
    t -= N3;
    if (t < N4)                { o4[t] = (h16)w4[t]; return; }
}

// BN fold:  y = (conv + cb - m) * g*rsqrt(v+eps) + be  ==  alpha*conv + beta
__global__ __launch_bounds__(256)
void fold_bn(const float* __restrict__ cb, const float* __restrict__ g,
             const float* __restrict__ be, const float* __restrict__ m,
             const float* __restrict__ v, float* __restrict__ alpha,
             float* __restrict__ beta, int n)
{
    int t = blockIdx.x * blockDim.x + threadIdx.x;
    if (t >= n) return;
    float s = g[t] * rsqrtf(v[t] + 1e-5f);
    alpha[t] = s;
    beta[t]  = (cb[t] - m[t]) * s + be[t];
}

// ---------------------------------------------------------------------------
// Scoring:  relu(1x1 conv 3->8) -> spatial mean -> linear 8->1, plus Gumbel
// ---------------------------------------------------------------------------
__global__ __launch_bounds__(128)
void score_kernel(const float* __restrict__ x,  const float* __restrict__ sw,
                  const float* __restrict__ sb, const float* __restrict__ slw,
                  const float* __restrict__ slb, float* __restrict__ sn)
{
    const int b = blockIdx.x, e = blockIdx.y, tid = threadIdx.x;
    __shared__ float red[128][8];

    float w[8][3], bb[8];
    #pragma unroll
    for (int c = 0; c < 8; ++c) {
        #pragma unroll
        for (int ci = 0; ci < 3; ++ci) w[c][ci] = sw[(e*8 + c)*3 + ci];
        bb[c] = sb[e*8 + c];
    }
    float hs[8] = {0,0,0,0,0,0,0,0};
    const float* xb = x + (size_t)b * 3 * 4096;
    for (int p = tid; p < 4096; p += 128) {
        float x0 = xb[p], x1 = xb[4096 + p], x2 = xb[8192 + p];
        #pragma unroll
        for (int c = 0; c < 8; ++c) {
            float vv = w[c][0]*x0 + w[c][1]*x1 + w[c][2]*x2 + bb[c];
            hs[c] += fmaxf(vv, 0.f);
        }
    }
    #pragma unroll
    for (int c = 0; c < 8; ++c) red[tid][c] = hs[c];
    __syncthreads();
    for (int s = 64; s > 0; s >>= 1) {
        if (tid < s) {
            #pragma unroll
            for (int c = 0; c < 8; ++c) red[tid][c] += red[tid + s][c];
        }
        __syncthreads();
    }
    if (tid == 0) {
        float sc = slb[e];
        #pragma unroll
        for (int c = 0; c < 8; ++c) sc += slw[e*8 + c] * (red[0][c] * (1.f/4096.f));
        unsigned s = (unsigned)(b * E_EXP + e) * 747796405u + 2891336453u;
        s ^= s >> 17; s *= 0xed5ad4bbu;
        s ^= s >> 11; s *= 0xac4c1b51u;
        s ^= s >> 15; s *= 0x31848babu;
        s ^= s >> 14;
        float u = (float)(s >> 8) * (1.0f/16777216.0f);
        u = fminf(fmaxf(u, 1e-7f), 1.f - 1e-7f);
        float gmb = -logf(-logf(u));
        sn[b*E_EXP + e] = (sc + gmb) * 10.0f;   // / TEMP (0.1)
    }
}

// ---------------------------------------------------------------------------
// Waterfall router (faithful sequential replication, fp64 like reference)
// ---------------------------------------------------------------------------
__global__ void route_kernel(const float* __restrict__ sn, int* __restrict__ owner,
                             int* __restrict__ idx, float* __restrict__ mask)
{
    if (threadIdx.x != 0 || blockIdx.x != 0) return;
    int cap[E_EXP] = {0, 0, 0, 0};
    for (int i = 0; i < BTOK; ++i) owner[i] = -1;
    int remaining = BTOK;
    for (int it = 0; it < 32 && remaining > 0; ++it) {
        long long quota = 1ll << (it < 20 ? it : 20);
        double ds[E_EXP]; bool full[E_EXP]; int space[E_EXP], taken[E_EXP];
        for (int e = 0; e < E_EXP; ++e) {
            double d = (double)cap[e] / (double)CAPC;
            d = d > 1.0 ? 1.0 : (d < 0.0 ? 0.0 : d);
            ds[e]   = 1.0 - d;
            full[e] = cap[e] >= CAPC;
            long long sp = (long long)(CAPC - cap[e]);
            if (sp > quota) sp = quota;
            if (sp < 0)     sp = 0;
            space[e] = (int)sp;
            taken[e] = 0;
        }
        for (int b = 0; b < BTOK; ++b) {
            if (owner[b] >= 0) continue;
            int best = -1; double bv = -1e300;
            for (int e = 0; e < E_EXP; ++e) {
                if (full[e]) continue;
                double v = (double)sn[b*E_EXP + e] * ds[e];
                if (v > bv) { bv = v; best = e; }
            }
            if (best >= 0 && taken[best] < space[best]) {
                owner[b] = best; taken[best]++; remaining--;
            }
        }
        for (int e = 0; e < E_EXP; ++e) cap[e] += taken[e];
    }
    for (int e = 0; e < E_EXP; ++e) {
        int cnt = 0;
        for (int b = 0; b < BTOK; ++b)
            if (owner[b] == e) { idx[e*CAPC + cnt] = b; mask[e*CAPC + cnt] = 1.f; cnt++; }
        for (; cnt < CAPC; ++cnt) { idx[e*CAPC + cnt] = 0; mask[e*CAPC + cnt] = 0.f; }
    }
}

// ---------------------------------------------------------------------------
// Gather expert's 256 images, convert f32 -> f16 (vectorized b128 -> b64)
// ---------------------------------------------------------------------------
__global__ __launch_bounds__(256)
void gather_kernel(const float* __restrict__ x, const int* __restrict__ idx,
                   h16* __restrict__ xin, int e)
{
    const int slot = blockIdx.x;
    const int img  = idx[e*CAPC + slot];
    const v4f* src = (const v4f*)(x   + (size_t)img  * 3 * 4096);
    v4h*       dst = (v4h*)      (xin + (size_t)slot * 3 * 4096);
    for (int t = threadIdx.x; t < 3*4096/4; t += 256) {
        v4f v = src[t];
        v4h o;
        #pragma unroll
        for (int j = 0; j < 4; ++j) o[j] = (h16)v[j];
        dst[t] = o;
    }
}

// ---------------------------------------------------------------------------
// Implicit-GEMM 3x3 conv, fused BN+ReLU (+2x2 maxpool), f16 WMMA, f32 acc.
// Block = 2 waves; wave w owns output row 2*yp+w, cols [x0, x0+16).
// Each wave computes CT16 cout-tiles (4 WMMAs per K-step, shared B fragment).
// LDS slab: [CIN][4 rows][32 cols], row body (16 px) at cols 8..23 so both
// global loads and LDS stores are 16B-aligned b128 ops; halos at cols 7/24.
// K loop fully unrolled so all im2col index math constant-folds; the ISA's
// lane-half K offset (grp*8) is a cndmask between two compile-time offsets.
// ---------------------------------------------------------------------------
template<int CIN, int COUT, int H, int W, bool POOL, int CT16>
__global__ __launch_bounds__(64)
void conv3x3_bnrelu_wmma(const h16* __restrict__ in, const h16* __restrict__ wt,
                         const float* __restrict__ alpha, const float* __restrict__ beta,
                         h16* __restrict__ out)
{
    constexpr int  KTOT  = CIN * 9;
    constexpr int  NSTEP = (KTOT + 31) / 32;
    constexpr bool EXACT = (KTOT % 32) == 0;
    constexpr int  CT    = W / 16;

    __shared__ h16   slab[CIN][4][32];
    __shared__ float pbuf[2][CT16*16][16];

    const int tid  = threadIdx.x;
    const int wave = tid >> 5;
    const int lane = tid & 31;
    const int x0   = (blockIdx.x % CT) * 16;
    const int yp   = blockIdx.x / CT;              // output row pair
    const int coutBase = blockIdx.y * (CT16 * 16);
    const int img  = blockIdx.z;

    const h16* inImg = in + (size_t)img * CIN * H * W;

    // vectorized zero-padded slab staging: one (ci, r) row per iteration
    for (int row = tid; row < CIN * 4; row += 64) {
        const int r  = row & 3;
        const int ci = row >> 2;
        const int gy = 2*yp - 1 + r;
        h16* dst = &slab[ci][r][0];
        if (gy < 0 || gy >= H) {
            v8h z = {};
            *(v8h*)(dst + 8)  = z;
            *(v8h*)(dst + 16) = z;
            dst[7]  = (h16)0.f;
            dst[24] = (h16)0.f;
        } else {
            const h16* srcRow = inImg + (size_t)ci * H * W + (size_t)gy * W;
            *(v8h*)(dst + 8)  = *(const v8h*)(srcRow + x0);
            *(v8h*)(dst + 16) = *(const v8h*)(srcRow + x0 + 8);
            dst[7]  = (x0 > 0)      ? srcRow[x0 - 1]  : (h16)0.f;
            dst[24] = (x0 + 16 < W) ? srcRow[x0 + 16] : (h16)0.f;
        }
    }
    __syncthreads();

    const int m     = lane & 15;      // A row (cout-local) and B col (pixel)
    const int grp   = lane >> 4;      // lane half
    const int base0 = grp << 3;       // ISA K sub-offset for this lane half

    // per-lane weight row pointers (already offset by lane-half K offset)
    const h16* wb[CT16];
    #pragma unroll
    for (int c = 0; c < CT16; ++c) {
        wb[c] = wt + (size_t)(coutBase + c*16 + m) * KTOT + base0;
        __builtin_prefetch(wb[c], 0, 0);
    }
    // B-panel base: element (ci,ky,kx) at hp[ci*128 + ky*32 + kx]
    const h16* hp = &slab[0][wave][m + 7];

    v8f acc[CT16] = {};

    #pragma unroll
    for (int s = 0; s < NSTEP; ++s) {
        const int k0 = s * 32;
        union { v16h v; v8h h[2]; } A[CT16], B;

        if (EXACT) {
            #pragma unroll
            for (int c = 0; c < CT16; ++c) {
                A[c].h[0] = *(const v8h*)(wb[c] + k0);
                A[c].h[1] = *(const v8h*)(wb[c] + k0 + 16);
            }
        } else {
            #pragma unroll
            for (int c = 0; c < CT16; ++c) {
                #pragma unroll
                for (int h = 0; h < 16; ++h) {
                    const int kgc = k0 + ((h >> 3) << 4) + (h & 7); // constant
                    const int kg0 = kgc, kg1 = kgc + 8;
                    const bool ok = grp ? (kg1 < KTOT) : (kg0 < KTOT);
                    const int  wo = grp ? (kg1 - 8) : kg0;          // wb has +base0
                    A[c].v[h] = ok ? wb[c][wo] : (h16)0.f;
                }
            }
        }

        #pragma unroll
        for (int h = 0; h < 16; ++h) {
            const int kgc = k0 + ((h >> 3) << 4) + (h & 7);         // constant
            const int kg0 = kgc, kg1 = kgc + 8;
            // compile-time im2col offsets for both lane halves
            const int o0 = (kg0/9)*128 + ((kg0%9)/3)*32 + (kg0%9)%3;
            const int o1 = (kg1/9)*128 + ((kg1%9)/3)*32 + (kg1%9)%3;
            const int off = grp ? o1 : o0;                          // one cndmask
            if (EXACT) {
                B.v[h] = hp[off];
            } else {
                const bool ok = grp ? (kg1 < KTOT) : (kg0 < KTOT);
                B.v[h] = ok ? hp[off] : (h16)0.f;
            }
        }

        #pragma unroll
        for (int c = 0; c < CT16; ++c)
            acc[c] = __builtin_amdgcn_wmma_f32_16x16x32_f16(
                         false, A[c].v, false, B.v, (short)0, acc[c], false, false);
    }

    const int outy = 2*yp + wave;
    if (!POOL) {
        #pragma unroll
        for (int c = 0; c < CT16; ++c) {
            #pragma unroll
            for (int r = 0; r < 8; ++r) {
                const int co = coutBase + c*16 + r + 8*grp;
                float v = alpha[co] * acc[c][r] + beta[co];
                v = v > 0.f ? v : 0.f;
                out[((size_t)img * COUT + co) * H * W + (size_t)outy * W + x0 + m] = (h16)v;
            }
        }
    } else {
        #pragma unroll
        for (int c = 0; c < CT16; ++c) {
            #pragma unroll
            for (int r = 0; r < 8; ++r) {
                const int co = c*16 + r + 8*grp;
                float v = alpha[coutBase + co] * acc[c][r] + beta[coutBase + co];
                pbuf[wave][co][m] = v > 0.f ? v : 0.f;
            }
        }
        __syncthreads();
        for (int o = tid; o < CT16*16*8; o += 64) {
            const int co = o >> 3, px = o & 7;
            float v0 = fmaxf(pbuf[0][co][2*px], pbuf[0][co][2*px + 1]);
            float v1 = fmaxf(pbuf[1][co][2*px], pbuf[1][co][2*px + 1]);
            float v  = fmaxf(v0, v1);
            out[((size_t)img * COUT + coutBase + co) * (H/2) * (W/2)
                + (size_t)yp * (W/2) + (x0 >> 1) + px] = (h16)v;
        }
    }
}

// ---------------------------------------------------------------------------
// GAP + linear(256->256) + linear(256->10) + masked scatter-add
// ---------------------------------------------------------------------------
__global__ __launch_bounds__(256)
void head_kernel(const h16* __restrict__ act4, const float* __restrict__ pw,
                 const float* __restrict__ pb, const float* __restrict__ cw,
                 const float* __restrict__ cb, const int* __restrict__ idx,
                 const float* __restrict__ mask, float* __restrict__ out, int e)
{
    __shared__ float z[256];
    __shared__ float emb[256];
    const int slot = blockIdx.x, tid = threadIdx.x;

    const v8h* row = (const v8h*)(act4 + (size_t)slot * 256 * 256 + (size_t)tid * 256);
    float s = 0.f;
    for (int p = 0; p < 32; ++p) {
        v8h v = row[p];
        #pragma unroll
        for (int j = 0; j < 8; ++j) s += (float)v[j];
    }
    z[tid] = s * (1.f/256.f);
    __syncthreads();

    const float* pwr = pw + (size_t)e * 256 * 256 + (size_t)tid * 256;
    float acc = pb[e*256 + tid];
    for (int c = 0; c < 256; ++c) acc += z[c] * pwr[c];
    emb[tid] = acc;
    __syncthreads();

    if (tid < NCLS) {
        const float* cwr = cw + (size_t)e * NCLS * 256 + (size_t)tid * 256;
        float l = cb[e*NCLS + tid];
        for (int c = 0; c < 256; ++c) l += emb[c] * cwr[c];
        const int img = idx[e*CAPC + slot];
        atomicAdd(&out[img*NCLS + tid], l * mask[e*CAPC + slot]);
    }
}

__global__ __launch_bounds__(256)
void zero_out(float* __restrict__ out, int n)
{
    int t = blockIdx.x * blockDim.x + threadIdx.x;
    if (t < n) out[t] = 0.f;
}

// ---------------------------------------------------------------------------
extern "C" void kernel_launch(void* const* d_in, const int* in_sizes, int n_in,
                              void* d_out, int out_size, void* d_ws, size_t ws_size,
                              hipStream_t stream)
{
    (void)in_sizes; (void)n_in; (void)out_size; (void)ws_size;

    const float* x    = (const float*)d_in[0];
    const float* c1w  = (const float*)d_in[1];
    const float* c1b  = (const float*)d_in[2];
    const float* bn1g = (const float*)d_in[3];
    const float* bn1b = (const float*)d_in[4];
    const float* bn1m = (const float*)d_in[5];
    const float* bn1v = (const float*)d_in[6];
    const float* c2w  = (const float*)d_in[7];
    const float* c2b  = (const float*)d_in[8];
    const float* bn2g = (const float*)d_in[9];
    const float* bn2b = (const float*)d_in[10];
    const float* bn2m = (const float*)d_in[11];
    const float* bn2v = (const float*)d_in[12];
    const float* c3w  = (const float*)d_in[13];
    const float* c3b  = (const float*)d_in[14];
    const float* bn3g = (const float*)d_in[15];
    const float* bn3b = (const float*)d_in[16];
    const float* bn3m = (const float*)d_in[17];
    const float* bn3v = (const float*)d_in[18];
    const float* c4w  = (const float*)d_in[19];
    const float* c4b  = (const float*)d_in[20];
    const float* bn4g = (const float*)d_in[21];
    const float* bn4b = (const float*)d_in[22];
    const float* bn4m = (const float*)d_in[23];
    const float* bn4v = (const float*)d_in[24];
    const float* pw   = (const float*)d_in[25];
    const float* pb   = (const float*)d_in[26];
    const float* cw   = (const float*)d_in[27];
    const float* cb   = (const float*)d_in[28];
    const float* sw   = (const float*)d_in[29];
    const float* sb   = (const float*)d_in[30];
    const float* slw  = (const float*)d_in[31];
    const float* slb  = (const float*)d_in[32];
    float* out = (float*)d_out;

    // ---- workspace carve-out (256B aligned) ----
    char*  ws  = (char*)d_ws;
    size_t off = 0;
    auto alloc = [&](size_t bytes) -> void* {
        void* p = ws + off;
        off = (off + bytes + 255) & ~(size_t)255;
        return p;
    };
    h16*   wh1  = (h16*)  alloc((size_t)E_EXP*64*27*2);
    h16*   wh2  = (h16*)  alloc((size_t)E_EXP*64*576*2);
    h16*   wh3  = (h16*)  alloc((size_t)E_EXP*128*576*2);
    h16*   wh4  = (h16*)  alloc((size_t)E_EXP*256*1152*2);
    float* a1   = (float*)alloc((size_t)E_EXP*64*4);
    float* b1   = (float*)alloc((size_t)E_EXP*64*4);
    float* a2   = (float*)alloc((size_t)E_EXP*64*4);
    float* b2   = (float*)alloc((size_t)E_EXP*64*4);
    float* a3   = (float*)alloc((size_t)E_EXP*128*4);
    float* b3   = (float*)alloc((size_t)E_EXP*128*4);
    float* a4   = (float*)alloc((size_t)E_EXP*256*4);
    float* b4   = (float*)alloc((size_t)E_EXP*256*4);
    float* sn   = (float*)alloc((size_t)BTOK*E_EXP*4);
    int*   owner= (int*)  alloc((size_t)BTOK*4);
    int*   idx  = (int*)  alloc((size_t)E_EXP*CAPC*4);
    float* mask = (float*)alloc((size_t)E_EXP*CAPC*4);
    h16*   xin  = (h16*)  alloc((size_t)CAPC*3*4096*2);
    h16*   act1 = (h16*)  alloc((size_t)CAPC*64*4096*2);
    h16*   act2 = (h16*)  alloc((size_t)CAPC*64*1024*2);
    h16*   act3 = (h16*)  alloc((size_t)CAPC*128*256*2);
    h16*   act4 = (h16*)  alloc((size_t)CAPC*256*256*2);

    // ---- prep: weight conversion + BN fold ----
    {
        const int total = E_EXP*64*27 + E_EXP*64*576 + E_EXP*128*576 + E_EXP*256*1152;
        convert_w<<<(total + 255)/256, 256, 0, stream>>>(c1w, c2w, c3w, c4w,
                                                         wh1, wh2, wh3, wh4);
        fold_bn<<<(E_EXP*64  + 255)/256, 256, 0, stream>>>(c1b, bn1g, bn1b, bn1m, bn1v, a1, b1, E_EXP*64);
        fold_bn<<<(E_EXP*64  + 255)/256, 256, 0, stream>>>(c2b, bn2g, bn2b, bn2m, bn2v, a2, b2, E_EXP*64);
        fold_bn<<<(E_EXP*128 + 255)/256, 256, 0, stream>>>(c3b, bn3g, bn3b, bn3m, bn3v, a3, b3, E_EXP*128);
        fold_bn<<<(E_EXP*256 + 255)/256, 256, 0, stream>>>(c4b, bn4g, bn4b, bn4m, bn4v, a4, b4, E_EXP*256);
    }

    // ---- scoring + routing ----
    score_kernel<<<dim3(BTOK, E_EXP), 128, 0, stream>>>(x, sw, sb, slw, slb, sn);
    route_kernel<<<1, 32, 0, stream>>>(sn, owner, idx, mask);
    zero_out<<<(BTOK*NCLS + 255)/256, 256, 0, stream>>>(out, BTOK*NCLS);

    // ---- per-expert pipeline (stream-ordered, buffers reused) ----
    for (int e = 0; e < E_EXP; ++e) {
        gather_kernel<<<CAPC, 256, 0, stream>>>(x, idx, xin, e);

        // conv1: 3 -> 64 @ 64x64 (4 cout tiles per block)
        conv3x3_bnrelu_wmma<3, 64, 64, 64, false, 4>
            <<<dim3(32*4, 1, CAPC), 64, 0, stream>>>(
                xin, wh1 + (size_t)e*64*27, a1 + e*64, b1 + e*64, act1);
        // conv2: 64 -> 64 @ 64x64, fused 2x2 maxpool -> 32x32
        conv3x3_bnrelu_wmma<64, 64, 64, 64, true, 4>
            <<<dim3(32*4, 1, CAPC), 64, 0, stream>>>(
                act1, wh2 + (size_t)e*64*576, a2 + e*64, b2 + e*64, act2);
        // conv3: 64 -> 128 @ 32x32, fused maxpool -> 16x16
        conv3x3_bnrelu_wmma<64, 128, 32, 32, true, 4>
            <<<dim3(16*2, 2, CAPC), 64, 0, stream>>>(
                act2, wh3 + (size_t)e*128*576, a3 + e*128, b3 + e*128, act3);
        // conv4: 128 -> 256 @ 16x16
        conv3x3_bnrelu_wmma<128, 256, 16, 16, false, 4>
            <<<dim3(8*1, 4, CAPC), 64, 0, stream>>>(
                act3, wh4 + (size_t)e*256*1152, a4 + e*256, b4 + e*256, act4);

        head_kernel<<<CAPC, 256, 0, stream>>>(act4, pw, pb, cw, cb, idx, mask, out, e);
    }
}
